// LinearizedContext_1357209665655
// MI455X (gfx1250) — matrix operand order repaired and checked by reading, not compiled, not run on hardware
//
#include <hip/hip_runtime.h>

// ---------------------------------------------------------------------------
// CDNA5 / gfx1250 fused scene-graph context model.
//   pack     -> all BERT weights fp32[K][N] -> bf16[N][Kpad] (transposed,
//               zero K-padded) so WMMA B-tiles are contiguous 32B/lane loads.
//   preproc  -> obj_pre (bf16, K padded 2376->2400)
//   bert x2  -> persistent per-image workgroup, all GEMMs via
//               v_wmma_f32_16x16x32_bf16, activations resident in LDS (256KB),
//               next-layer weight prefetch via global_prefetch_b8
//   classify -> obj_dists2
//   nms      -> greedy class labels (sequential, single block)
// ---------------------------------------------------------------------------

typedef __attribute__((ext_vector_type(16))) __bf16 v16bf;
typedef __attribute__((ext_vector_type(8)))  float  v8f;

union V16 { v16bf v; unsigned u[8]; __bf16 h[16]; };

#define T_TOK 64
#define HID   256
#define LDX   2400   // 2376 padded to a multiple of 32 (75 K-steps)
#define NCLS  151
#define NOBJ  512

// round-to-nearest-even float -> bf16
__device__ inline __bf16 f2bf(float f) {
  unsigned u = __builtin_bit_cast(unsigned, f);
  unsigned r = (u + 0x7FFFu + ((u >> 16) & 1u)) >> 16;
  unsigned short s = (unsigned short)r;
  return __builtin_bit_cast(__bf16, s);
}

__device__ inline v8f vzero8() { v8f z = {0.f,0.f,0.f,0.f,0.f,0.f,0.f,0.f}; return z; }

__device__ inline v8f wmma_bf16(v16bf a, v16bf b, v8f c) {
  // (neg_a, A, neg_b, B, c_mod, C, reuse_a, reuse_b)
  return __builtin_amdgcn_wmma_f32_16x16x32_bf16(false, a, false, b, (short)0, c, false, false);
}

// A tile 16x32 bf16, row-major source (row stride ld elems), tile origin p.
// lane r(0-15)/r+16 -> row r; VGPR v -> K = hi*16 + 2*v (contiguous, matches B loaders)
__device__ inline v16bf load_a(const __bf16* p, int ld, int lane) {
  int r = lane & 15, hi = lane >> 4;
  const __bf16* row = p + r * ld;
  V16 a;
#pragma unroll
  for (int v = 0; v < 8; ++v) {
    int kb = (hi << 4) + (v << 1);
    a.u[v] = *reinterpret_cast<const unsigned*>(row + kb);
  }
  return a.v;
}

// B tile 32x16 where B[k][n] = src[n*ld + k] (source row-major [N][K];
// each lane reads 32 contiguous bytes -> vectorizes to global_load_b128).
__device__ inline v16bf load_b_tr(const __bf16* p, int ld, int lane) {
  int c = lane & 15, hk = (lane >> 4) << 4;
  const __bf16* row = p + c * ld + hk;
  V16 b;
#pragma unroll
  for (int v = 0; v < 8; ++v) b.u[v] = *reinterpret_cast<const unsigned*>(row + 2 * v);
  return b.v;
}

// B tile 32x16 where B[k][n] = src[k*ld + n] (bf16 LDS source, strided rows)
__device__ inline v16bf load_b_col(const __bf16* p, int ld, int lane) {
  int c = lane & 15, hk = (lane >> 4) << 4;
  V16 b;
#pragma unroll
  for (int v = 0; v < 8; ++v) {
    b.h[2*v]   = p[(hk + 2*v)     * ld + c];
    b.h[2*v+1] = p[(hk + 2*v + 1) * ld + c];
  }
  return b.v;
}

__device__ inline float gelu_tanh(float x) {
  float x3 = x * x * x;
  return 0.5f * x * (1.f + tanhf(0.7978845608028654f * (x + 0.044715f * x3)));
}

// Packed (pre-transposed bf16) weights + fp32 biases/LN params.
struct BertP {
  const __bf16 *w_in, *wq, *wk, *wv, *wo, *w1, *w2;   // [N][Kpad] bf16
  const float  *b_in, *bq, *bk, *bv, *bo, *b1, *b2;
  const float  *ln1_g, *ln1_b, *ln2_g, *ln2_b;
};

// ---------------------------------------------------------------------------
// Weight pack: src fp32 [L][K][N] -> dst bf16 [L][N][Kpad], zero pad k>=K.
// Writes fully coalesced; strided reads are L2-resident (192 MB L2).
// ---------------------------------------------------------------------------
__global__ void pack_kernel(const float* __restrict__ src, __bf16* __restrict__ dst,
                            int K, int N, int Kpad) {
  size_t l = blockIdx.y;
  const float* s = src + l * (size_t)K * N;
  __bf16* d = dst + l * (size_t)N * Kpad;
  int total = N * Kpad;
  for (int idx = blockIdx.x * blockDim.x + threadIdx.x; idx < total;
       idx += gridDim.x * blockDim.x) {
    int n = idx / Kpad, k = idx - n * Kpad;
    d[idx] = f2bf(k < K ? s[(size_t)k * N + n] : 0.f);
  }
}

// LayerNorm h[64][256] (fp32) -> zb (bf16); one wave handles 8 rows,
// per-row reduction via full-wave shfl_xor (wave32).
__device__ inline void layernorm_rows(const float* __restrict__ hbuf, __bf16* __restrict__ zb,
                                      const float* g, const float* b, int wave, int lane) {
#pragma unroll
  for (int r8 = 0; r8 < 8; ++r8) {
    int row = wave * 8 + r8;
    const float* x = hbuf + row * HID;
    float s = 0.f, s2 = 0.f;
#pragma unroll
    for (int i = 0; i < 8; ++i) { float v = x[lane + 32 * i]; s += v; s2 += v * v; }
#pragma unroll
    for (int m = 16; m >= 1; m >>= 1) { s += __shfl_xor(s, m, 32); s2 += __shfl_xor(s2, m, 32); }
    float mean = s * (1.f / HID);
    float var  = s2 * (1.f / HID) - mean * mean;
    float rs   = rsqrtf(var + 1e-6f);
#pragma unroll
    for (int i = 0; i < 8; ++i) {
      int cc = lane + 32 * i;
      zb[row * HID + cc] = f2bf((x[cc] - mean) * rs * g[cc] + b[cc]);
    }
  }
}

// ---------------------------------------------------------------------------
// Fused 12-layer BERT: one block (8 wave32 waves) per image.
// Dynamic LDS layout (262144 B total):
//   [0      , 65536) : h      fp32 [64][256]  residual stream
//   [65536  , 98304) : zb     bf16 [64][256]  LN out / attn out (o)
//   [98304  ,262144) : union:
//       attn: qs|ks|vs bf16 [64][256] x3  + probs bf16 [8][64][64]
//       ffn : f bf16 [64][1024]
// ---------------------------------------------------------------------------
__global__ __launch_bounds__(256)
void bert_kernel(const __bf16* __restrict__ xin, BertP P, float* __restrict__ ctx_out) {
  extern __shared__ char smem[];
  float*  hbuf = (float*)smem;
  __bf16* zb   = (__bf16*)(smem + 65536);
  __bf16* qs   = (__bf16*)(smem + 98304);
  __bf16* ksb  = qs  + T_TOK * HID;
  __bf16* vsb  = ksb + T_TOK * HID;
  __bf16* prob = vsb + T_TOK * HID;             // 8 x 64 x 64 (wave-private)
  __bf16* ffn  = (__bf16*)(smem + 98304);       // 64 x 1024

  int tid = threadIdx.x, lane = tid & 31, wave = tid >> 5;
  int img = blockIdx.x;
  const __bf16* X = xin + (size_t)img * T_TOK * LDX;

  // ---- input projection: h = X @ w_in + b_in  (Kpad = 2400, pad pre-zeroed)
  for (int t = wave; t < 64; t += 8) {
    int m0 = (t >> 4) << 4, n0 = (t & 15) << 4;
    v8f c = vzero8();
    for (int k0 = 0; k0 < LDX; k0 += 32) {
      v16bf a = load_a(X + (size_t)m0 * LDX + k0, LDX, lane);
      v16bf b = load_b_tr(P.w_in + (size_t)n0 * LDX + k0, LDX, lane);
      c = wmma_bf16(a, b, c);
    }
    int col = lane & 15, rb = (lane >> 4) << 3;
    float bias = P.b_in[n0 + col];
#pragma unroll
    for (int v = 0; v < 8; ++v) hbuf[(m0 + rb + v) * HID + n0 + col] = c[v] + bias;
  }
  __syncthreads();

  for (int L = 0; L < 12; ++L) {
    const __bf16* wq = P.wq + (size_t)L * HID * HID;
    const __bf16* wk = P.wk + (size_t)L * HID * HID;
    const __bf16* wv = P.wv + (size_t)L * HID * HID;
    const __bf16* wo = P.wo + (size_t)L * HID * HID;
    const __bf16* w1 = P.w1 + (size_t)L * HID * 1024;   // [1024][256]
    const __bf16* w2 = P.w2 + (size_t)L * 1024 * HID;   // [256][1024]
    const float* bq = P.bq + L * HID;  const float* bk = P.bk + L * HID;
    const float* bv = P.bv + L * HID;  const float* bo = P.bo + L * HID;
    const float* b1 = P.b1 + L * 1024; const float* b2 = P.b2 + L * HID;
    const float* g1 = P.ln1_g + L * HID; const float* e1 = P.ln1_b + L * HID;
    const float* g2 = P.ln2_g + L * HID; const float* e2 = P.ln2_b + L * HID;

    // Warm the WGP cache with next layer's weights while this layer computes
    // (L2->WGP latency hiding; emits global_prefetch_b8, RT speculative).
    if (L + 1 < 12) {
      const char* nxt0 = (const char*)(P.wq + (size_t)(L + 1) * HID * HID);
      const char* nxt1 = (const char*)(P.wk + (size_t)(L + 1) * HID * HID);
      const char* nxt2 = (const char*)(P.wv + (size_t)(L + 1) * HID * HID);
      const char* nxt3 = (const char*)(P.wo + (size_t)(L + 1) * HID * HID);
      const char* nxt4 = (const char*)(P.w1 + (size_t)(L + 1) * HID * 1024);
      const char* nxt5 = (const char*)(P.w2 + (size_t)(L + 1) * 1024 * HID);
      size_t o = (size_t)tid * 256;
      __builtin_prefetch(nxt0 + o, 0, 1);
      __builtin_prefetch(nxt1 + o, 0, 1);
      __builtin_prefetch(nxt2 + o, 0, 1);
      __builtin_prefetch(nxt3 + o, 0, 1);
      __builtin_prefetch(nxt4 + o, 0, 1);
      __builtin_prefetch(nxt5 + o, 0, 1);
    }

    layernorm_rows(hbuf, zb, g1, e1, wave, lane);
    __syncthreads();

    // ---- QKV: {q,k,v} = zb @ W + b  (3 GEMMs, 192 tiles over 8 waves) ----
    for (int t = wave; t < 192; t += 8) {
      int which = t >> 6, tt = t & 63;
      int m0 = (tt >> 4) << 4, n0 = (tt & 15) << 4;
      const __bf16* W  = (which == 0) ? wq : ((which == 1) ? wk : wv);
      const float*  Bi = (which == 0) ? bq : ((which == 1) ? bk : bv);
      __bf16* Dst      = (which == 0) ? qs : ((which == 1) ? ksb : vsb);
      v8f c = vzero8();
      for (int k0 = 0; k0 < HID; k0 += 32) {
        v16bf a = load_a(zb + m0 * HID + k0, HID, lane);
        v16bf b = load_b_tr(W + (size_t)n0 * HID + k0, HID, lane);
        c = wmma_bf16(a, b, c);
      }
      int col = lane & 15, rb = (lane >> 4) << 3;
      float bias = Bi[n0 + col];
#pragma unroll
      for (int v = 0; v < 8; ++v) Dst[(m0 + rb + v) * HID + n0 + col] = f2bf(c[v] + bias);
    }
    __syncthreads();

    // ---- attention: wave w owns head w (head_dim = 32 = one WMMA K) ----
    {
      int hb = wave * 32;
      __bf16* pp = prob + (size_t)wave * T_TOK * T_TOK;
      const float SC = 0.17677669529663687f;   // 1/sqrt(32)
      for (int mt = 0; mt < 4; ++mt) {
        v16bf a = load_a(qs + (mt * 16) * HID + hb, HID, lane);
        v8f s[4];
#pragma unroll
        for (int nt = 0; nt < 4; ++nt) {
          v16bf b = load_b_tr(ksb + (nt * 16) * HID + hb, HID, lane);  // K^T tile
          s[nt] = wmma_bf16(a, b, vzero8());
        }
        // softmax over 64 keys: C-layout rows live on a 16-lane half group
#pragma unroll
        for (int v = 0; v < 8; ++v) {
          float e[4];
#pragma unroll
          for (int nt = 0; nt < 4; ++nt) e[nt] = s[nt][v] * SC;
          float mx = fmaxf(fmaxf(e[0], e[1]), fmaxf(e[2], e[3]));
#pragma unroll
          for (int m = 8; m >= 1; m >>= 1) mx = fmaxf(mx, __shfl_xor(mx, m, 32));
          float sum = 0.f;
#pragma unroll
          for (int nt = 0; nt < 4; ++nt) { e[nt] = expf(e[nt] - mx); sum += e[nt]; }
#pragma unroll
          for (int m = 8; m >= 1; m >>= 1) sum += __shfl_xor(sum, m, 32);
          float inv = 1.f / sum;
          int row = mt * 16 + ((lane >> 4) << 3) + v;
          int col = lane & 15;
#pragma unroll
          for (int nt = 0; nt < 4; ++nt) pp[row * T_TOK + nt * 16 + col] = f2bf(e[nt] * inv);
        }
      }
      // o(head) = probs @ V   -> write into zb (LN buffer is dead now)
      for (int mt = 0; mt < 4; ++mt) {
#pragma unroll
        for (int d = 0; d < 2; ++d) {
          v8f c = vzero8();
#pragma unroll
          for (int kt = 0; kt < 2; ++kt) {
            v16bf a = load_a(pp + (mt * 16) * T_TOK + kt * 32, T_TOK, lane);
            v16bf b = load_b_col(vsb + (kt * 32) * HID + hb + d * 16, HID, lane);
            c = wmma_bf16(a, b, c);
          }
          int col = lane & 15, rb = (lane >> 4) << 3;
#pragma unroll
          for (int v = 0; v < 8; ++v)
            zb[(mt * 16 + rb + v) * HID + hb + d * 16 + col] = f2bf(c[v]);
        }
      }
    }
    __syncthreads();

    // ---- O-projection: h += o @ wo + bo ----
    for (int t = wave; t < 64; t += 8) {
      int m0 = (t >> 4) << 4, n0 = (t & 15) << 4;
      v8f c = vzero8();
      for (int k0 = 0; k0 < HID; k0 += 32) {
        v16bf a = load_a(zb + m0 * HID + k0, HID, lane);
        v16bf b = load_b_tr(wo + (size_t)n0 * HID + k0, HID, lane);
        c = wmma_bf16(a, b, c);
      }
      int col = lane & 15, rb = (lane >> 4) << 3;
      float bias = bo[n0 + col];
#pragma unroll
      for (int v = 0; v < 8; ++v) hbuf[(m0 + rb + v) * HID + n0 + col] += c[v] + bias;
    }
    __syncthreads();

    layernorm_rows(hbuf, zb, g2, e2, wave, lane);
    __syncthreads();

    // ---- FFN1: f = gelu(zb @ w1 + b1)  [64][1024] ----
    for (int t = wave; t < 256; t += 8) {
      int m0 = (t & 3) << 4, n0 = (t >> 2) << 4;
      v8f c = vzero8();
      for (int k0 = 0; k0 < HID; k0 += 32) {
        v16bf a = load_a(zb + m0 * HID + k0, HID, lane);
        v16bf b = load_b_tr(w1 + (size_t)n0 * HID + k0, HID, lane);
        c = wmma_bf16(a, b, c);
      }
      int col = lane & 15, rb = (lane >> 4) << 3;
      float bias = b1[n0 + col];
#pragma unroll
      for (int v = 0; v < 8; ++v)
        ffn[(m0 + rb + v) * 1024 + n0 + col] = f2bf(gelu_tanh(c[v] + bias));
    }
    __syncthreads();

    // ---- FFN2: h += f @ w2 + b2 ----
    for (int t = wave; t < 64; t += 8) {
      int m0 = (t >> 4) << 4, n0 = (t & 15) << 4;
      v8f c = vzero8();
      for (int k0 = 0; k0 < 1024; k0 += 32) {
        v16bf a = load_a(ffn + m0 * 1024 + k0, 1024, lane);
        v16bf b = load_b_tr(w2 + (size_t)n0 * 1024 + k0, 1024, lane);
        c = wmma_bf16(a, b, c);
      }
      int col = lane & 15, rb = (lane >> 4) << 3;
      float bias = b2[n0 + col];
#pragma unroll
      for (int v = 0; v < 8; ++v) hbuf[(m0 + rb + v) * HID + n0 + col] += c[v] + bias;
    }
    __syncthreads();
  }

  float* out = ctx_out + (size_t)img * T_TOK * HID;
  for (int idx = tid; idx < T_TOK * HID; idx += 256) out[idx] = hbuf[idx];
}

// ---------------------------------------------------------------------------
// Preprocessing: softmax-embed + BN(center_size) + pos MLP -> obj_pre bf16
// ---------------------------------------------------------------------------
__global__ void preproc_kernel(const float* __restrict__ fmaps, const float* __restrict__ logits,
                               const float* __restrict__ boxes, const float* __restrict__ embW,
                               const float* __restrict__ bn_g, const float* __restrict__ bn_b,
                               const float* __restrict__ pos_w, const float* __restrict__ pos_b,
                               __bf16* __restrict__ xpre) {
  int i = blockIdx.x, tid = threadIdx.x;
  __shared__ float sm[NCLS + 1];
  __shared__ float red[256];

  float mx = -1e30f;
  for (int c = tid; c < NCLS; c += 256) { float v = logits[i * NCLS + c]; sm[c] = v; mx = fmaxf(mx, v); }
  red[tid] = mx; __syncthreads();
  for (int s = 128; s >= 1; s >>= 1) { if (tid < s) red[tid] = fmaxf(red[tid], red[tid + s]); __syncthreads(); }
  float M = red[0]; __syncthreads();

  float sum = 0.f;
  for (int c = tid; c < NCLS; c += 256) { float e = expf(sm[c] - M); sm[c] = e; sum += e; }
  red[tid] = sum; __syncthreads();
  for (int s = 128; s >= 1; s >>= 1) { if (tid < s) red[tid] += red[tid + s]; __syncthreads(); }
  float inv = 1.f / red[0]; __syncthreads();
  for (int c = tid; c < NCLS; c += 256) sm[c] *= inv;
  __syncthreads();

  for (int c = tid; c < 2048; c += 256)
    xpre[(size_t)i * LDX + c] = f2bf(fmaps[(size_t)i * 2048 + c]);

  for (int j = tid; j < 200; j += 256) {
    float acc = 0.f;
    for (int c = 0; c < NCLS; ++c) acc += sm[c] * embW[c * 200 + j];
    xpre[(size_t)i * LDX + 2048 + j] = f2bf(acc);
  }

  float x1 = boxes[i * 4 + 0], y1 = boxes[i * 4 + 1], x2 = boxes[i * 4 + 2], y2 = boxes[i * 4 + 3];
  float w = x2 - x1 + 1.f, hgt = y2 - y1 + 1.f;
  float cs[4] = { x1 + 0.5f * w, y1 + 0.5f * hgt, w, hgt };
  const float invs = 0.9999950000374997f;   // 1/sqrt(1 + 1e-5)
  float bn[4];
#pragma unroll
  for (int k = 0; k < 4; ++k) bn[k] = cs[k] * invs * bn_g[k] + bn_b[k];
  for (int j = tid; j < 128; j += 256) {
    float acc = pos_b[j];
#pragma unroll
    for (int k = 0; k < 4; ++k) acc += bn[k] * pos_w[k * 128 + j];
    xpre[(size_t)i * LDX + 2248 + j] = f2bf(fmaxf(acc, 0.f));
  }
  if (tid < LDX - 2376) xpre[(size_t)i * LDX + 2376 + tid] = f2bf(0.f);  // K padding
}

// ---------------------------------------------------------------------------
// obj_dists2 = ctx @ cls_w + cls_b + obj_logits  (tiny; scalar per-output)
// ---------------------------------------------------------------------------
__global__ void classify_kernel(const float* __restrict__ ctx, const float* __restrict__ cls_w,
                                const float* __restrict__ cls_b, const float* __restrict__ logits,
                                float* __restrict__ dists) {
  int i = blockIdx.x, j = threadIdx.x;
  if (j >= NCLS) return;
  float acc = cls_b[j] + logits[i * NCLS + j];
  const float* c = ctx + (size_t)i * HID;
  for (int k = 0; k < HID; ++k) acc += c[k] * cls_w[k * NCLS + j];
  dists[(size_t)i * NCLS + j] = acc;
}

// ---------------------------------------------------------------------------
// Greedy NMS labeling (sequential, 512 iterations). IoU computed on the fly.
// ---------------------------------------------------------------------------
__global__ void nms_kernel(const float* __restrict__ dists, const float* __restrict__ boxes,
                           float* __restrict__ dsm, int* __restrict__ preds_i,
                           float* __restrict__ preds_f) {
  __shared__ float sval[512];
  __shared__ int   sidx[512];
  __shared__ int   labs[512];
  int tid = threadIdx.x;
  labs[tid] = 0;

  // per-row softmax; background class zeroed
  {
    const float* r = dists + (size_t)tid * NCLS;
    float* o = dsm + (size_t)tid * NCLS;
    float mx = -1e30f;
    for (int c = 0; c < NCLS; ++c) mx = fmaxf(mx, r[c]);
    float sum = 0.f;
    for (int c = 0; c < NCLS; ++c) { float e = expf(r[c] - mx); o[c] = e; sum += e; }
    float inv = 1.f / sum;
    for (int c = 0; c < NCLS; ++c) o[c] *= inv;
    o[0] = 0.f;
  }
  __syncthreads();

  for (int it = 0; it < NOBJ; ++it) {
    float best = -1e30f; int bidx = 0;
    for (int j = tid; j < NOBJ * NCLS; j += 512) {
      float v = dsm[j];
      if (v > best) { best = v; bidx = j; }
    }
    sval[tid] = best; sidx[tid] = bidx;
    __syncthreads();
    for (int s = 256; s >= 1; s >>= 1) {
      if (tid < s) {
        float ov = sval[tid + s]; int oi = sidx[tid + s];
        if (ov > sval[tid] || (ov == sval[tid] && oi < sidx[tid])) { sval[tid] = ov; sidx[tid] = oi; }
      }
      __syncthreads();
    }
    int bb = sidx[0] / NCLS, cc = sidx[0] % NCLS;
    if (tid == 0) labs[bb] = cc;
    {
      const float* A  = boxes + ((size_t)bb  * NCLS + cc) * 4;
      const float* Bx = boxes + ((size_t)tid * NCLS + cc) * 4;
      float iw = fminf(A[2], Bx[2]) - fmaxf(A[0], Bx[0]) + 1.f;
      float ih = fminf(A[3], Bx[3]) - fmaxf(A[1], Bx[1]) + 1.f;
      iw = fmaxf(iw, 0.f); ih = fmaxf(ih, 0.f);
      float inter = iw * ih;
      float aa = (A[2] - A[0] + 1.f) * (A[3] - A[1] + 1.f);
      float ab = (Bx[2] - Bx[0] + 1.f) * (Bx[3] - Bx[1] + 1.f);
      float iou = inter / (aa + ab - inter);
      if (iou >= 0.3f) dsm[(size_t)tid * NCLS + cc] = 0.f;
    }
    __syncthreads();
    if (tid < NCLS) dsm[(size_t)bb * NCLS + tid] = -1.f;
    __syncthreads();
  }
  preds_i[tid] = labs[tid];
  preds_f[tid] = (float)labs[tid];
}

// Replace the embed slice of obj_pre with obj_embed_in_edge_w[pred]
__global__ void edge_embed_kernel(const int* __restrict__ preds, const float* __restrict__ embW,
                                  __bf16* __restrict__ xpre) {
  int i = blockIdx.x, cl = preds[i];
  for (int j = threadIdx.x; j < 200; j += blockDim.x)
    xpre[(size_t)i * LDX + 2048 + j] = f2bf(embW[cl * 200 + j]);
}

// ---------------------------------------------------------------------------
// Host. Input order: top-level dict insertion order; params pytree flattened
// in JAX order (sorted keys, depth-first):
//  0 obj_fmaps 1 obj_logits 2 box_priors 3 boxes_per_cls 4 im_inds
//  5 bn_b 6 bn_g 7 cls_b 8 cls_w
//  9 edge.b_in 10..25 edge.layers{b1,b2,bk,bo,bq,bv,ln1_b,ln1_g,ln2_b,ln2_g,
//                                 w1,w2,wk,wo,wq,wv} 26 edge.w_in
// 27 obj.b_in  28..43 obj.layers(same order)         44 obj.w_in
// 45 obj_embed_in_edge_w 46 obj_embed_w 47 pos_b 48 pos_w
// ---------------------------------------------------------------------------
extern "C" void kernel_launch(void* const* d_in, const int* in_sizes, int n_in,
                              void* d_out, int out_size, void* d_ws, size_t ws_size,
                              hipStream_t stream) {
  (void)in_sizes; (void)n_in; (void)out_size; (void)ws_size;

  const float* obj_fmaps  = (const float*)d_in[0];
  const float* obj_logits = (const float*)d_in[1];
  const float* box_priors = (const float*)d_in[2];
  const float* boxes_cls  = (const float*)d_in[3];
  const float* bn_b  = (const float*)d_in[5];
  const float* bn_g  = (const float*)d_in[6];
  const float* cls_b = (const float*)d_in[7];
  const float* cls_w = (const float*)d_in[8];
  const float* emb_edge_w = (const float*)d_in[45];
  const float* emb_obj_w  = (const float*)d_in[46];
  const float* pos_b = (const float*)d_in[47];
  const float* pos_w = (const float*)d_in[48];

  // ---- workspace layout ----
  char* ws = (char*)d_ws;
  size_t off = 0;
  auto salloc = [&](size_t bytes) { size_t o = off; off += (bytes + 255) & ~(size_t)255; return o; };
  __bf16* xpre  = (__bf16*)(ws + salloc((size_t)NOBJ * LDX * 2));
  float*  ctx   = (float*) (ws + salloc((size_t)NOBJ * HID * 4));
  float*  dsm   = (float*) (ws + salloc((size_t)NOBJ * NCLS * 4));
  int*    preds = (int*)   (ws + salloc((size_t)NOBJ * 4));

  // Packed bf16 weights for both BERTs (~40 MB). Pack once per launch;
  // entire parameter set is L2-resident on MI455X (192 MB L2).
  struct Idx { int b_in, base, w_in; };
  Idx bidx[2] = { {27, 28, 44},   // obj bert
                  { 9, 10, 26} }; // edge bert
  BertP bp[2];
  for (int b = 0; b < 2; ++b) {
    int base = bidx[b].base;
    BertP& p = bp[b];
    p.b_in  = (const float*)d_in[bidx[b].b_in];
    p.b1    = (const float*)d_in[base + 0];  p.b2    = (const float*)d_in[base + 1];
    p.bk    = (const float*)d_in[base + 2];  p.bo    = (const float*)d_in[base + 3];
    p.bq    = (const float*)d_in[base + 4];  p.bv    = (const float*)d_in[base + 5];
    p.ln1_b = (const float*)d_in[base + 6];  p.ln1_g = (const float*)d_in[base + 7];
    p.ln2_b = (const float*)d_in[base + 8];  p.ln2_g = (const float*)d_in[base + 9];
    const float* w1f   = (const float*)d_in[base + 10];
    const float* w2f   = (const float*)d_in[base + 11];
    const float* wkf   = (const float*)d_in[base + 12];
    const float* wof   = (const float*)d_in[base + 13];
    const float* wqf   = (const float*)d_in[base + 14];
    const float* wvf   = (const float*)d_in[base + 15];
    const float* w_inf = (const float*)d_in[bidx[b].w_in];

    __bf16* w_inP = (__bf16*)(ws + salloc((size_t)HID * LDX * 2));            // [256][2400]
    __bf16* wqP   = (__bf16*)(ws + salloc((size_t)12 * HID * HID * 2));
    __bf16* wkP   = (__bf16*)(ws + salloc((size_t)12 * HID * HID * 2));
    __bf16* wvP   = (__bf16*)(ws + salloc((size_t)12 * HID * HID * 2));
    __bf16* woP   = (__bf16*)(ws + salloc((size_t)12 * HID * HID * 2));
    __bf16* w1P   = (__bf16*)(ws + salloc((size_t)12 * 1024 * HID * 2));      // [1024][256]
    __bf16* w2P   = (__bf16*)(ws + salloc((size_t)12 * HID * 1024 * 2));      // [256][1024]
    p.w_in = w_inP; p.wq = wqP; p.wk = wkP; p.wv = wvP; p.wo = woP; p.w1 = w1P; p.w2 = w2P;

    pack_kernel<<<dim3((HID * LDX + 255) / 256, 1), 256, 0, stream>>>(w_inf, w_inP, 2376, HID, LDX);
    pack_kernel<<<dim3((HID * HID + 255) / 256, 12), 256, 0, stream>>>(wqf, wqP, HID, HID, HID);
    pack_kernel<<<dim3((HID * HID + 255) / 256, 12), 256, 0, stream>>>(wkf, wkP, HID, HID, HID);
    pack_kernel<<<dim3((HID * HID + 255) / 256, 12), 256, 0, stream>>>(wvf, wvP, HID, HID, HID);
    pack_kernel<<<dim3((HID * HID + 255) / 256, 12), 256, 0, stream>>>(wof, woP, HID, HID, HID);
    pack_kernel<<<dim3((1024 * HID + 255) / 256, 12), 256, 0, stream>>>(w1f, w1P, HID, 1024, HID);
    pack_kernel<<<dim3((HID * 1024 + 255) / 256, 12), 256, 0, stream>>>(w2f, w2P, 1024, HID, 1024);
  }

  float* out_dists = (float*)d_out;             // [512*151]
  float* out_preds = out_dists + NOBJ * NCLS;   // [512] (labels as float)
  float* out_edge  = out_preds + NOBJ;          // [512*256]

  const size_t BERT_LDS = 262144;               // 256 KB of the 320 KB WGP LDS

  preproc_kernel<<<NOBJ, 256, 0, stream>>>(obj_fmaps, obj_logits, box_priors, emb_obj_w,
                                           bn_g, bn_b, pos_w, pos_b, xpre);
  bert_kernel<<<8, 256, BERT_LDS, stream>>>(xpre, bp[0], ctx);
  classify_kernel<<<NOBJ, 160, 0, stream>>>(ctx, cls_w, cls_b, obj_logits, out_dists);
  nms_kernel<<<1, 512, 0, stream>>>(out_dists, boxes_cls, dsm, preds, out_preds);
  edge_embed_kernel<<<NOBJ, 256, 0, stream>>>(preds, emb_edge_w, xpre);
  bert_kernel<<<8, 256, BERT_LDS, stream>>>(xpre, bp[1], out_edge);
}
